// GraphLearner_67954972557816
// MI455X (gfx1250) — compile-verified, matrix operand order checked by reading, not run on hardware
//
#include <hip/hip_runtime.h>
#include <hip/hip_bf16.h>
#include <math.h>

#define BB 2
#define TT 256
#define NN 512
#define HH 64

typedef __attribute__((ext_vector_type(16))) _Float16 v16h;
typedef __attribute__((ext_vector_type(8)))  _Float16 v8h;
typedef __attribute__((ext_vector_type(8)))  float    v8f;
typedef __attribute__((ext_vector_type(4)))  unsigned int v4u;
typedef __attribute__((ext_vector_type(8)))  int          v8i;
typedef __attribute__((ext_vector_type(4)))  int          v4i;

// Tensor Data Mover: builtin exists on both probed toolchains, with different
// arity (5 args on ROCm 7.2 / clang-22, 6 args on clang-23).
#if defined(__HIP_DEVICE_COMPILE__) && defined(__clang__)
#  if __has_builtin(__builtin_amdgcn_tensor_load_to_lds)
#    define HAVE_TDM 1
#  endif
#endif
#ifndef HAVE_TDM
#  define HAVE_TDM 0
#endif

__device__ __forceinline__ float softplusf(float x) {
    return (x > 20.f) ? x : log1pf(expf(x));
}

// ---------------------------------------------------------------------------
// Kernel 1: per-(b,n) time-series stats -> dyn[b,n,4] = {mean, std, last, mr}
// ---------------------------------------------------------------------------
__global__ void k_stats(const float* __restrict__ x, const float* __restrict__ mask,
                        float* __restrict__ dyn) {
    int idx = blockIdx.x * blockDim.x + threadIdx.x;     // 0 .. B*N-1
    if (idx >= BB * NN) return;
    int b = idx / NN, n = idx % NN;
    const float* xb = x    + (size_t)b * TT * NN + n;
    const float* mb = mask + (size_t)b * TT * NN + n;
    float Sx = 0.f, Sxx = 0.f, Cobs = 0.f, Smask = 0.f;
    int lastT = 0;
    for (int t = 0; t < TT; ++t) {
        float xv = xb[(size_t)t * NN];
        float mv = mb[(size_t)t * NN];
        float obs = 1.f - mv;
        Sx   += xv * obs;
        Sxx  += xv * xv * obs;
        Cobs += obs;
        Smask += mv;
        if (obs > 0.5f && t > 0) lastT = t;              // argmax(t*obs) semantics
    }
    float cnt  = fmaxf(Cobs, 1.f);
    float mean = Sx / cnt;
    float ssq  = fmaxf(Sxx - 2.f * mean * Sx + mean * mean * Cobs, 0.f);
    float stdv = sqrtf(ssq / cnt + 1e-6f);
    float last = xb[(size_t)lastT * NN];
    float mr   = Smask / (float)TT;
    float* d = dyn + (size_t)idx * 4;
    d[0] = mean; d[1] = stdv; d[2] = last; d[3] = mr;
}

// ---------------------------------------------------------------------------
// Kernel 2: swizzle Wa=We1[128:192], Wp=We1[192:256] (f32) into f16 buffer in
// exact per-lane WMMA B-operand order:
//   wswz[ ((mat*4+g)*2+chunk)*512 + lane*16 + e ]
// ---------------------------------------------------------------------------
__global__ void k_wswz(const float* __restrict__ We1, _Float16* __restrict__ wswz) {
    int id = blockIdx.x * blockDim.x + threadIdx.x;      // 0 .. 8191
    if (id >= 8192) return;
    int e     = id & 15;
    int lane  = (id >> 4) & 31;
    int chunk = (id >> 9) & 1;
    int g     = (id >> 10) & 3;
    int mat   = (id >> 12) & 1;                          // 0 = Wa, 1 = Wp
    int half  = lane >> 4;
    int col   = g * 16 + (lane & 15);                    // B-operand lane = out col
    int kk    = ((e < 8) ? (half * 8 + e) : (16 + half * 8 + (e - 8))) + chunk * 32;
    int row   = 128 + mat * 64 + kk;                     // We1 row index
    wswz[id]  = (_Float16)We1[row * HH + col];
}

// ---------------------------------------------------------------------------
// Kernel 3: per-node MLP: feats(12) -> h(64); emit h16, hn, pre_i^T, pre_j
// one block of 64 threads per (b,n).  pre_i stored transposed [c][node] so the
// edge kernel can read a D-tile half's 8 row-bias values as one v8f load.
// ---------------------------------------------------------------------------
__global__ void k_mlp(const float* __restrict__ dyn, const float* __restrict__ statc,
                      const float* __restrict__ W1, const float* __restrict__ b1,
                      const float* __restrict__ W2, const float* __restrict__ b2,
                      const float* __restrict__ We1,
                      _Float16* __restrict__ h16, float* __restrict__ hn,
                      float* __restrict__ prei_t, float* __restrict__ prej) {
    int node = blockIdx.x;                               // b*N + n
    int k = threadIdx.x;                                 // 0..63
    int n = node % NN;
    __shared__ float feats[12], h1[HH], h2[HH], red[HH];
    if (k < 4)       feats[k] = dyn[(size_t)node * 4 + k];
    else if (k < 12) feats[k] = statc[(size_t)n * 8 + (k - 4)];
    __syncthreads();
    float a = b1[k];
    #pragma unroll
    for (int j = 0; j < 12; ++j) a += feats[j] * W1[j * HH + k];
    h1[k] = fmaxf(a, 0.f);
    __syncthreads();
    float a2 = b2[k];
    for (int c = 0; c < HH; ++c) a2 += h1[c] * W2[c * HH + k];
    float hv = fmaxf(a2, 0.f);
    h2[k] = hv;
    red[k] = hv * hv;
    __syncthreads();
    for (int s = 32; s > 0; s >>= 1) {
        if (k < s) red[k] += red[k + s];
        __syncthreads();
    }
    float nrm = fmaxf(sqrtf(red[0]), 1e-12f);
    hn[(size_t)node * HH + k]  = hv / nrm;
    h16[(size_t)node * HH + k] = (_Float16)hv;
    float pi = 0.f, pj = 0.f;
    for (int c = 0; c < HH; ++c) {
        float hc = h2[c];
        pi += hc * We1[c * HH + k];            // Wi = We1 rows [0,64)
        pj += hc * We1[(HH + c) * HH + k];     // Wj = We1 rows [64,128)
    }
    prei_t[(size_t)k * (BB * NN) + node] = pi;           // transposed
    prej[(size_t)node * HH + k] = pj;
}

// ---------------------------------------------------------------------------
// Kernel 4 (WMMA): 256-thread blocks = 8 waves; wave -> (b, j, 32-row i block
// = two 16-row tiles).  Weights staged once per block into LDS via the Tensor
// Data Mover (cooperative-copy fallback); operands read with ds_load_b128.
// 32 v_wmma_f32_16x16x32_f16 per wave; epilogue reads pre_i^T as v8f, then
// bias+ReLU+We2 contraction via half-wave shfl_xor reduction.
// ---------------------------------------------------------------------------
__global__ void __launch_bounds__(256)
k_edge(const _Float16* __restrict__ h16, const _Float16* __restrict__ wswz,
       const float* __restrict__ prei_t, const float* __restrict__ prej,
       const float* __restrict__ be1, const float* __restrict__ We2,
       const float* __restrict__ be2, float* __restrict__ edge) {
    __shared__ _Float16 smem_w[8192];                    // 16 KB swizzled Wa|Wp

#if HAVE_TDM
    if (threadIdx.x < 32) {                              // wave 0 issues the TDM op
        unsigned lds_off = (unsigned)(unsigned long long)(const void*)smem_w;
        unsigned long long ga = (unsigned long long)wswz;
        // D# group0: count=1 (valid), lds_addr, 57-bit global addr, type=2
        v4u g0;
        g0[0] = 1u;
        g0[1] = lds_off;
        g0[2] = (unsigned)ga;
        g0[3] = ((unsigned)(ga >> 32) & 0x01FFFFFFu) | (2u << 30);
        // D# group1: data_size=8B; 1-D tensor/tile of 2048 elements (16 KB)
        v8i g1;
        g1[0] = 0x30000;                 // data_size = 3 (8 bytes)
        g1[1] = (int)(2048u << 16);      // tensor_dim0[15:0] in bits[31:16]
        g1[2] = (int)(1u << 16);         // tensor_dim1 = 1
        g1[3] = (int)(2048u << 16);      // tile_dim0 = 2048
        g1[4] = 1;                       // tile_dim1 = 1
        g1[5] = 2048;                    // tensor_dim0_stride[31:0]
        g1[6] = (int)(2048u << 16);      // tensor_dim1_stride[15:0]
        g1[7] = 0;
        v4i gz = {0, 0, 0, 0};
#  if defined(__clang_major__) && (__clang_major__ >= 23)
        v8i gpad = {0, 0, 0, 0, 0, 0, 0, 0};
        __builtin_amdgcn_tensor_load_to_lds(g0, g1, gz, gz, gpad, 0);
#  else
        __builtin_amdgcn_tensor_load_to_lds(g0, g1, gz, gz, 0);
#  endif
        __builtin_amdgcn_s_wait_tensorcnt((short)0);
    }
    __syncthreads();
#else
    {
        const uint4* src = (const uint4*)wswz;
        uint4* dst = (uint4*)smem_w;
        for (int t = threadIdx.x; t < 1024; t += 256) dst[t] = src[t];
        __syncthreads();
    }
#endif

    int wid  = blockIdx.x * 8 + (threadIdx.x >> 5);
    int b    = wid >> 13;                // N * (N/32) = 8192 waves per batch
    int rem  = wid & 8191;
    int j    = rem >> 4;
    int i0   = (rem & 15) << 5;          // 32-row i block
    int lane = threadIdx.x & 31;
    int m    = lane & 15;
    int half = lane >> 4;

    // h_j row segments (shared by both i tiles):
    //   seg s: K = s*16 + 8*half .. +7 ; chunk = s>>1, element offset = (s&1)*8
    const _Float16* hj_row = h16 + (size_t)(b * NN + j) * HH;
    v8h js[4];
    #pragma unroll
    for (int s = 0; s < 4; ++s)
        js[s] = *(const v8h*)(hj_row + s * 16 + 8 * half);

    v16h A[2][2], P[2][2];               // [tile][chunk]
    #pragma unroll
    for (int t = 0; t < 2; ++t) {
        const _Float16* hi_row = h16 + (size_t)(b * NN + i0 + t * 16 + m) * HH;
        #pragma unroll
        for (int s = 0; s < 4; ++s) {
            v8h is = *(const v8h*)(hi_row + s * 16 + 8 * half);
            int c  = s >> 1;
            int eo = (s & 1) * 8;
            #pragma unroll
            for (int e = 0; e < 8; ++e) {
                float av = (float)is[e], bv = (float)js[s][e];
                A[t][c][eo + e] = (_Float16)fabsf(av - bv);
                P[t][c][eo + e] = (_Float16)(av * bv);
            }
        }
    }

    float edgeacc[2][8];
    #pragma unroll
    for (int t = 0; t < 2; ++t)
        #pragma unroll
        for (int v = 0; v < 8; ++v) edgeacc[t][v] = 0.f;

    const _Float16* wl = smem_w + lane * 16;             // this lane's operand slot
    #pragma unroll
    for (int g = 0; g < 4; ++g) {                        // 4 groups of 16 channels
        int col = g * 16 + m;
        v16h Wa0 = *(const v16h*)(wl + ((0 * 4 + g) * 2 + 0) * 512);
        v16h Wa1 = *(const v16h*)(wl + ((0 * 4 + g) * 2 + 1) * 512);
        v16h Wp0 = *(const v16h*)(wl + ((1 * 4 + g) * 2 + 0) * 512);
        v16h Wp1 = *(const v16h*)(wl + ((1 * 4 + g) * 2 + 1) * 512);
        float bias = prej[(size_t)(b * NN + j) * HH + col] + be1[col];
        float w2   = We2[col];
        #pragma unroll
        for (int t = 0; t < 2; ++t) {
            v8f acc = {};
            acc = __builtin_amdgcn_wmma_f32_16x16x32_f16(false, A[t][0], false, Wa0, (short)0, acc, false, false);
            acc = __builtin_amdgcn_wmma_f32_16x16x32_f16(false, A[t][1], false, Wa1, (short)0, acc, false, false);
            acc = __builtin_amdgcn_wmma_f32_16x16x32_f16(false, P[t][0], false, Wp0, (short)0, acc, false, false);
            acc = __builtin_amdgcn_wmma_f32_16x16x32_f16(false, P[t][1], false, Wp1, (short)0, acc, false, false);
            // 8 row-bias values for this D-tile half: one aligned v8f load
            v8f pv = *(const v8f*)(prei_t + (size_t)col * (BB * NN)
                                   + b * NN + i0 + t * 16 + 8 * half);
            #pragma unroll
            for (int v = 0; v < 8; ++v) {
                float val = fmaxf(acc[v] + pv[v] + bias, 0.f);
                edgeacc[t][v] += val * w2;
            }
        }
    }

    float bb = be2[0];
    #pragma unroll
    for (int t = 0; t < 2; ++t) {
        #pragma unroll
        for (int v = 0; v < 8; ++v) {
            float s = edgeacc[t][v];
            s += __shfl_xor(s, 1, 32);
            s += __shfl_xor(s, 2, 32);
            s += __shfl_xor(s, 4, 32);
            s += __shfl_xor(s, 8, 32);                   // sum over 16-lane half
            if (m == 0) {
                int i = i0 + t * 16 + v + half * 8;
                edge[((size_t)(b * NN + i)) * NN + j] = s + bb;
            }
        }
    }
}

// ---------------------------------------------------------------------------
// Kernel 5: combine logits (edge + prior + cos + self-loop), softmax per row
// ---------------------------------------------------------------------------
__global__ void k_logits_softmax(const float* __restrict__ edge, const float* __restrict__ hn,
                                 const float* __restrict__ coords,
                                 const float* __restrict__ p_self, const float* __restrict__ p_prior,
                                 const float* __restrict__ p_sim,  const float* __restrict__ p_temp,
                                 float* __restrict__ adj) {
    int row = blockIdx.x;                                // b*N + i
    int b = row / NN, i = row % NN;
    int tid = threadIdx.x;
    __shared__ float hni[HH];
    __shared__ float ci[8];
    __shared__ float red[256];
    if (tid < HH) hni[tid] = hn[(size_t)row * HH + tid];
    if (tid < 8)  ci[tid]  = coords[i * 8 + tid];
    __syncthreads();
    float sp_self  = softplusf(p_self[0]);
    float sp_prior = softplusf(p_prior[0]);
    float sp_sim   = softplusf(p_sim[0]);
    float temp     = softplusf(p_temp[0]) + 1e-4f;
    float lg[2];
    #pragma unroll
    for (int q = 0; q < 2; ++q) {
        int j = tid + q * 256;
        const float* hnj = hn + (size_t)(b * NN + j) * HH;
        float cosv = 0.f;
        for (int c = 0; c < HH; ++c) cosv += hni[c] * hnj[c];
        float d2 = 0.f;
        #pragma unroll
        for (int c = 0; c < 8; ++c) { float df = ci[c] - coords[j * 8 + c]; d2 += df * df; }
        float dist  = sqrtf(fmaxf(d2, 1e-12f));
        float prior = 1.f / (1.f + dist);
        float off   = (i == j) ? 0.f : 1.f;
        float el    = edge[(size_t)row * NN + j];
        float logit = el * off + sp_prior * prior * off + sp_sim * cosv * off
                    + (1.f - off) * sp_self;
        lg[q] = logit / temp;
    }
    red[tid] = fmaxf(lg[0], lg[1]);
    __syncthreads();
    for (int s = 128; s > 0; s >>= 1) {
        if (tid < s) red[tid] = fmaxf(red[tid], red[tid + s]);
        __syncthreads();
    }
    float mx = red[0];
    __syncthreads();
    float e0 = expf(lg[0] - mx), e1 = expf(lg[1] - mx);
    red[tid] = e0 + e1;
    __syncthreads();
    for (int s = 128; s > 0; s >>= 1) {
        if (tid < s) red[tid] += red[tid + s];
        __syncthreads();
    }
    float inv = 1.f / red[0];
    adj[(size_t)row * NN + tid]       = e0 * inv;
    adj[(size_t)row * NN + tid + 256] = e1 * inv;
}

// ---------------------------------------------------------------------------
// Kernel 6: symmetrize + row-normalize -> d_out
// ---------------------------------------------------------------------------
__global__ void k_sym(const float* __restrict__ adj, float* __restrict__ out) {
    int row = blockIdx.x;                                // b*N + i
    int b = row / NN, i = row % NN;
    int tid = threadIdx.x;
    __shared__ float red[256];
    int j0 = tid, j1 = tid + 256;
    float s0 = 0.5f * (adj[(size_t)row * NN + j0] + adj[((size_t)(b * NN + j0)) * NN + i]);
    float s1 = 0.5f * (adj[(size_t)row * NN + j1] + adj[((size_t)(b * NN + j1)) * NN + i]);
    red[tid] = s0 + s1;
    __syncthreads();
    for (int s = 128; s > 0; s >>= 1) {
        if (tid < s) red[tid] += red[tid + s];
        __syncthreads();
    }
    float inv = 1.f / fmaxf(red[0], 1e-6f);
    out[(size_t)row * NN + j0] = s0 * inv;
    out[(size_t)row * NN + j1] = s1 * inv;
}

// ---------------------------------------------------------------------------
extern "C" void kernel_launch(void* const* d_in, const int* in_sizes, int n_in,
                              void* d_out, int out_size, void* d_ws, size_t ws_size,
                              hipStream_t stream) {
    (void)in_sizes; (void)n_in; (void)out_size; (void)ws_size;
    const float* x      = (const float*)d_in[0];
    const float* mask   = (const float*)d_in[1];
    const float* statc  = (const float*)d_in[2];
    const float* coords = (const float*)d_in[3];
    const float* W1     = (const float*)d_in[4];
    const float* b1     = (const float*)d_in[5];
    const float* W2     = (const float*)d_in[6];
    const float* b2     = (const float*)d_in[7];
    const float* We1    = (const float*)d_in[8];
    const float* be1    = (const float*)d_in[9];
    const float* We2    = (const float*)d_in[10];
    const float* be2    = (const float*)d_in[11];
    const float* p_self = (const float*)d_in[12];
    const float* p_prior= (const float*)d_in[13];
    const float* p_sim  = (const float*)d_in[14];
    const float* p_temp = (const float*)d_in[15];
    float* out = (float*)d_out;

    // workspace layout (all 256B aligned)
    char* ws = (char*)d_ws;
    auto take = [&](size_t bytes) -> char* {
        char* p = ws;
        ws += (bytes + 255) & ~(size_t)255;
        return p;
    };
    float*    dyn    = (float*)   take((size_t)BB * NN * 4 * sizeof(float));
    _Float16* h16    = (_Float16*)take((size_t)BB * NN * HH * sizeof(_Float16));
    _Float16* wswz   = (_Float16*)take((size_t)8192 * sizeof(_Float16));
    float*    prei_t = (float*)   take((size_t)HH * BB * NN * sizeof(float));
    float*    prej   = (float*)   take((size_t)BB * NN * HH * sizeof(float));
    float*    hnbuf  = (float*)   take((size_t)BB * NN * HH * sizeof(float));
    float*    edge   = (float*)   take((size_t)BB * NN * NN * sizeof(float));
    float*    adj    = (float*)   take((size_t)BB * NN * NN * sizeof(float));

    k_stats<<<(BB * NN + 255) / 256, 256, 0, stream>>>(x, mask, dyn);
    k_wswz <<<8192 / 256, 256, 0, stream>>>(We1, wswz);
    k_mlp  <<<BB * NN, HH, 0, stream>>>(dyn, statc, W1, b1, W2, b2, We1,
                                        h16, hnbuf, prei_t, prej);
    k_edge <<<BB * NN * (NN / 32) / 8, 256, 0, stream>>>(h16, wswz, prei_t, prej,
                                                         be1, We2, be2, edge);
    k_logits_softmax<<<BB * NN, 256, 0, stream>>>(edge, hnbuf, coords,
                                                  p_self, p_prior, p_sim, p_temp, adj);
    k_sym  <<<BB * NN, 256, 0, stream>>>(adj, out);
}
